// LICv1_36103495090264
// MI455X (gfx1250) — matrix operand order repaired and checked by reading, not compiled, not run on hardware
//
#include <hip/hip_runtime.h>
#include <math.h>

typedef __attribute__((ext_vector_type(16))) _Float16 v16h;
typedef __attribute__((ext_vector_type(8)))  _Float16 v8h;
typedef __attribute__((ext_vector_type(8)))  float    v8f;

#define NB   4096   // total batch
#define BPB  16     // batch rows per block
#define E    64     // embedding
#define LL   200
#define LS   20
#define NDT  128

// ---- WMMA fragment loaders (CDNA5 wave32 layouts, CDNA5 ISA §7.12.2) ----
// A (f16 16x32) from row-major [M][K]: lane<16 -> K runs {0..15}; lane>=16 -> {16..31}.
// Each run is 16 contiguous halves -> two ds_load_b128.
__device__ __forceinline__ v16h lda_f16(const _Float16* p, int stride, int ko, int lane) {
  int m = lane & 15, hi = (lane >> 4) & 1;
  v8h lo  = *(const v8h*)(p + m * stride + ko + hi * 16);
  v8h hi8 = *(const v8h*)(p + m * stride + ko + 8 + hi * 16);
  return __builtin_shufflevector(lo, hi8, 0,1,2,3,4,5,6,7,8,9,10,11,12,13,14,15);
}
// B (f16 32x16) from COLUMN-major storage (element [K][N] at p[N*strideK + K]).
// Fragment element e has K = hi*16 + e -> 16 contiguous halves -> two ds_load_b128.
__device__ __forceinline__ v16h ldb_f16(const _Float16* p, int strideK, int ko, int no, int lane) {
  int n = lane & 15, hi = (lane >> 4) & 1;
  const _Float16* q = p + (no + n) * strideK + ko + (hi << 4);
  v8h b0 = *(const v8h*)q;
  v8h b1 = *(const v8h*)(q + 8);
  return __builtin_shufflevector(b0, b1, 0,1,2,3,4,5,6,7,8,9,10,11,12,13,14,15);
}

__global__ __launch_bounds__(512, 1) void clockrec_main(
    const int* __restrict__ items_l, const int* __restrict__ dts_l, const unsigned char* __restrict__ mask_l,
    const int* __restrict__ items_s, const int* __restrict__ dts_s, const unsigned char* __restrict__ mask_s,
    const int* __restrict__ pos_it, const int* __restrict__ neg_it,
    const float* __restrict__ item_table,
    const float* __restrict__ fW1, const float* __restrict__ fb1,
    const float* __restrict__ fW2, const float* __restrict__ fb2,
    const float* __restrict__ g_dt, const float* __restrict__ g_cW, const float* __restrict__ g_cb,
    const float* __restrict__ g_kW, const float* __restrict__ g_kb,
    const float* __restrict__ g_vW, const float* __restrict__ g_vb, const float* __restrict__ g_tau,
    const float* __restrict__ e_dt, const float* __restrict__ e_cW, const float* __restrict__ e_cb,
    const float* __restrict__ e_kW, const float* __restrict__ e_kb,
    const float* __restrict__ e_vW, const float* __restrict__ e_vb, const float* __restrict__ e_tau,
    float* __restrict__ out, float* __restrict__ lam_ws)
{
  __shared__ alignas(16) _Float16 Qh[2][BPB][E];     // [cand][b][e] candidate embeddings (f16, row-major)
  __shared__ alignas(16) _Float16 Wb[192 * E];       // reusable weight buffer, COLUMN-major [N][K]
  __shared__ alignas(16) _Float16 KWQ[2][2][BPB][E]; // [unit][cand][b][e] : k_W @ q
  __shared__ alignas(16) _Float16 UB[2][2][BPB][E];  // Ubar (weighted token sum) then U (after @v_W)
  __shared__ float Hb[2][BPB][E];                    // fuse hidden
  __shared__ float dtsc[2][NDT][2];                  // sin/cos per dt bucket
  __shared__ float cWs[2][2][E];
  __shared__ float cbs[2][E];
  __shared__ float vbs[2][E];
  __shared__ float kbq[2][2][BPB];                   // k_b . q
  __shared__ float fb1s[E];
  __shared__ float fw2s[E];
  __shared__ float tauS[2];
  __shared__ float lamS[2][BPB];
  __shared__ float fb2s;

  const int tid  = threadIdx.x;
  const int lane = tid & 31;
  const int wv   = tid >> 5;          // 16 waves
  const int bb   = blockIdx.x * BPB;  // first global batch row

  // ---------------- Phase 0: staging ----------------
  for (int i = tid; i < 2 * NDT; i += 512) {
    int u = i >> 7, d = i & 127;
    float ang = (6.283185307179586f / 127.0f) * (float)d;
    dtsc[u][d][0] = __sinf(ang);
    dtsc[u][d][1] = __cosf(ang);
  }
  for (int i = tid; i < 2 * 2 * E; i += 512) {
    int u = i >> 7, r = (i >> 6) & 1, c = i & 63;
    cWs[u][r][c] = (u ? e_cW : g_cW)[r * E + c];
  }
  for (int i = tid; i < 2 * E; i += 512) {
    int u = i >> 6, c = i & 63;
    cbs[u][c] = (u ? e_cb : g_cb)[c];
    vbs[u][c] = (u ? e_vb : g_vb)[c];
  }
  for (int i = tid; i < E; i += 512) { fb1s[i] = fb1[i]; fw2s[i] = fW2[i]; }
  if (tid == 0) fb2s = fb2[0];
  if (tid < 2) tauS[tid] = log1pf(expf((tid ? e_tau : g_tau)[0])) + 1e-6f;
  // candidate embeddings (f16)
  for (int i = tid; i < 2 * BPB * E; i += 512) {
    int c = i >> 10, m = (i >> 6) & 15, e = i & 63;
    int it = (c ? neg_it : pos_it)[bb + m];
    Qh[c][m][e] = (_Float16)item_table[(size_t)it * E + e];
  }
  // k_b . q
  if (tid < 2 * 2 * BPB) {
    int u = tid >> 5, c = (tid >> 4) & 1, m = tid & 15;
    int it = (c ? neg_it : pos_it)[bb + m];
    const float* q = item_table + (size_t)it * E;
    const float* kb = u ? e_kb : g_kb;
    float s = 0.f;
    for (int e2 = 0; e2 < E; ++e2) s += kb[e2] * q[e2];
    kbq[u][c][m] = s;
  }
  // Wb <- B for KWQ GEMM: B[K=n][N=e] = kW[e*64+n]; col-major [N=e][K=n] = kW[e*64+n] -> DIRECT copy
  for (int i = tid; i < 2 * E * E; i += 512) {
    int u = i >> 12, r = i & 4095;
    Wb[u * 4096 + r] = (_Float16)((u ? e_kW : g_kW)[r]);
  }
  __syncthreads();

  // ---------------- Phase 1: KWQ = Q @ k_W^T (4 waves, one GEMM each) ----------------
  if (wv < 4) {
    int u = wv >> 1, c = wv & 1;
    const _Float16* pa = &Qh[c][0][0];
    const _Float16* pb = &Wb[u * 4096];
    v16h a0 = lda_f16(pa, E, 0, lane);
    v16h a1 = lda_f16(pa, E, 32, lane);
    int n = lane & 15, hi = (lane >> 4) & 1;
#pragma unroll
    for (int j = 0; j < 4; ++j) {
      v8f acc = {};
      v16h b0 = ldb_f16(pb, E, 0, j * 16, lane);
      v16h b1 = ldb_f16(pb, E, 32, j * 16, lane);
      acc = __builtin_amdgcn_wmma_f32_16x16x32_f16(false, a0, false, b0, (short)0, acc, false, false);
      acc = __builtin_amdgcn_wmma_f32_16x16x32_f16(false, a1, false, b1, (short)0, acc, false, false);
#pragma unroll
      for (int r = 0; r < 8; ++r) KWQ[u][c][r + 8 * hi][j * 16 + n] = (_Float16)acc[r];
    }
  }
  __syncthreads();

  // ---------------- Phase 2: token pass, online softmax ----------------
  // Wave wv = batch row. Lanes 0-15 own token t, lanes 16-31 own token t+1;
  // each lane owns 4 embedding elements -> b128 gathers, 4-level half-reductions.
  {
    const int m = wv, gb = bb + m;
    const int half = lane >> 4, sl = lane & 15, e0 = sl * 4;
#pragma unroll 1
    for (int u = 0; u < 2; ++u) {
      const int L = u ? LS : LL;
      const int* its = u ? items_s : items_l;
      const int* dts = u ? dts_s : dts_l;
      const unsigned char* msk = u ? mask_s : mask_l;
      const float* dtt = u ? e_dt : g_dt;
      const int base = gb * L;
      float invt = 1.0f / tauS[u];
      float kp[4], kn[4], c0v[4], c1v[4], cbv[4];
#pragma unroll
      for (int j = 0; j < 4; ++j) {
        kp[j] = (float)KWQ[u][0][m][e0 + j] * invt;
        kn[j] = (float)KWQ[u][1][m][e0 + j] * invt;
        c0v[j] = cWs[u][0][e0 + j];
        c1v[j] = cWs[u][1][e0 + j];
        cbv[j] = cbs[u][e0 + j];
      }
      float bp = kbq[u][0][m] * invt, bn = kbq[u][1][m] * invt;
      float mp = -INFINITY, dp = 0.f, up[4] = {0.f, 0.f, 0.f, 0.f};
      float mn = -INFINITY, dn = 0.f, un[4] = {0.f, 0.f, 0.f, 0.f};
      for (int t = 0; t < L; t += 2) {
        const int tt = t + half;
        int it = its[base + tt];
        int dv = dts[base + tt];
        bool mk = msk[base + tt] != 0;
        // prefetch the item row ~8 tokens ahead (global_prefetch_b8)
        int tpf = tt + 8;
        if (tpf < L) {
          int itp = its[base + tpf];
          __builtin_prefetch((const void*)(item_table + (size_t)itp * E + e0), 0, 3);
        }
        float4 iv  = ((const float4*)(item_table + (size_t)it * E))[sl];
        float4 dvv = ((const float4*)(dtt + dv * E))[sl];
        float sn = dtsc[u][dv][0], cs = dtsc[u][dv][1];
        float tok[4];
        tok[0] = iv.x + dvv.x + sn * c0v[0] + cs * c1v[0] + cbv[0];
        tok[1] = iv.y + dvv.y + sn * c0v[1] + cs * c1v[1] + cbv[1];
        tok[2] = iv.z + dvv.z + sn * c0v[2] + cs * c1v[2] + cbv[2];
        tok[3] = iv.w + dvv.w + sn * c0v[3] + cs * c1v[3] + cbv[3];
        float lp = tok[0] * kp[0] + tok[1] * kp[1] + tok[2] * kp[2] + tok[3] * kp[3];
        float ln = tok[0] * kn[0] + tok[1] * kn[1] + tok[2] * kn[2] + tok[3] * kn[3];
#pragma unroll
        for (int o = 8; o > 0; o >>= 1) { lp += __shfl_xor(lp, o, 32); ln += __shfl_xor(ln, o, 32); }
        lp = mk ? lp + bp : -INFINITY;
        ln = mk ? ln + bn : -INFINITY;
        float lp_o = __shfl_xor(lp, 16, 32);
        float ln_o = __shfl_xor(ln, 16, 32);
        float lpe = half ? lp_o : lp, lpo = half ? lp : lp_o;   // even/odd token logits (wave-uniform)
        float lne = half ? ln_o : ln, lno = half ? ln : ln_o;
        // pos candidate
        float mpn = fmaxf(mp, fmaxf(lpe, lpo));
        if (mpn > -INFINITY) {
          float sc = __expf(mp - mpn);                 // exp(-inf)=0 handles first token
          float we = __expf(lpe - mpn);                // masked logits (-inf) -> 0
          float wo = __expf(lpo - mpn);
          dp = dp * sc + we + wo;
          float wown = half ? wo : we;
#pragma unroll
          for (int j = 0; j < 4; ++j) up[j] = up[j] * sc + wown * tok[j];
          mp = mpn;
        }
        // neg candidate
        float mnn = fmaxf(mn, fmaxf(lne, lno));
        if (mnn > -INFINITY) {
          float sc = __expf(mn - mnn);
          float we = __expf(lne - mnn);
          float wo = __expf(lno - mnn);
          dn = dn * sc + we + wo;
          float wown = half ? wo : we;
#pragma unroll
          for (int j = 0; j < 4; ++j) un[j] = un[j] * sc + wown * tok[j];
          mn = mnn;
        }
      }
      // combine halves (elements split across lane halves by token parity)
#pragma unroll
      for (int j = 0; j < 4; ++j) {
        up[j] += __shfl_xor(up[j], 16, 32);
        un[j] += __shfl_xor(un[j], 16, 32);
      }
      if (half == 0) {
        float rp = 1.0f / dp, rn = 1.0f / dn;
#pragma unroll
        for (int j = 0; j < 4; ++j) {
          UB[u][0][m][e0 + j] = (_Float16)(up[j] * rp);
          UB[u][1][m][e0 + j] = (_Float16)(un[j] * rn);
        }
      }
    }
  }
  __syncthreads();

  // ---------------- Phase 3: U = Ubar @ v_W + v_b (Wb <- v_W, col-major) ----------------
  for (int i = tid; i < 2 * E * E; i += 512) {
    int u = i >> 12, n = (i >> 6) & 63, k = i & 63;
    Wb[u * 4096 + n * E + k] = (_Float16)((u ? e_vW : g_vW)[k * E + n]);
  }
  __syncthreads();
  if (wv < 4) {
    int u = wv >> 1, c = wv & 1;
    const _Float16* pa = &UB[u][c][0][0];
    const _Float16* pb = &Wb[u * 4096];
    v16h a0 = lda_f16(pa, E, 0, lane);     // A captured in regs before in-place overwrite
    v16h a1 = lda_f16(pa, E, 32, lane);
    int n = lane & 15, hi = (lane >> 4) & 1;
#pragma unroll
    for (int j = 0; j < 4; ++j) {
      v8f acc = {};
      v16h b0 = ldb_f16(pb, E, 0, j * 16, lane);
      v16h b1 = ldb_f16(pb, E, 32, j * 16, lane);
      acc = __builtin_amdgcn_wmma_f32_16x16x32_f16(false, a0, false, b0, (short)0, acc, false, false);
      acc = __builtin_amdgcn_wmma_f32_16x16x32_f16(false, a1, false, b1, (short)0, acc, false, false);
#pragma unroll
      for (int r = 0; r < 8; ++r)
        UB[u][c][r + 8 * hi][j * 16 + n] = (_Float16)(acc[r] + vbs[u][j * 16 + n]);
    }
  }
  __syncthreads();

  // ---------------- Phase 4: fuse MLP h = relu([q,u_g,u_e] @ fuse_W1 + b1) ----------------
  // Wb col-major [N=n][K] with strideK=192: Wb[n*192+k] = fW1[k*64+n]
  for (int i = tid; i < 192 * E; i += 512) {
    int n = i / 192, k = i - n * 192;
    Wb[i] = (_Float16)fW1[k * E + n];
  }
  __syncthreads();
  if (wv < 2) {
    int c = wv;
    v16h af[6];
#pragma unroll
    for (int ch = 0; ch < 6; ++ch) {
      int ko = ch * 32;
      const _Float16* basep = (ko < 64) ? &Qh[c][0][0] : (ko < 128) ? &UB[0][c][0][0] : &UB[1][c][0][0];
      af[ch] = lda_f16(basep, E, ko & 63, lane);
    }
    int n = lane & 15, hi = (lane >> 4) & 1;
#pragma unroll
    for (int j = 0; j < 4; ++j) {
      v8f acc = {};
#pragma unroll
      for (int ch = 0; ch < 6; ++ch) {
        v16h b = ldb_f16(&Wb[0], 192, ch * 32, j * 16, lane);
        acc = __builtin_amdgcn_wmma_f32_16x16x32_f16(false, af[ch], false, b, (short)0, acc, false, false);
      }
#pragma unroll
      for (int r = 0; r < 8; ++r)
        Hb[c][r + 8 * hi][j * 16 + n] = fmaxf(acc[r] + fb1s[j * 16 + n], 0.0f);
    }
  }
  __syncthreads();

  // ---------------- Phase 5: lambda, blend, score ----------------
  if (tid < 2 * BPB) {
    int c = tid >> 4, m = tid & 15;
    float z = fb2s;
    for (int e2 = 0; e2 < E; ++e2) z += Hb[c][m][e2] * fw2s[e2];
    lamS[c][m] = 1.0f / (1.0f + expf(-z));
  }
  __syncthreads();
  if (tid < 2 * BPB) {
    int c = tid >> 4, m = tid & 15;
    float lam = lamS[c][m], s = 0.f;
    for (int e2 = 0; e2 < E; ++e2) {
      float uu = lam * (float)UB[0][c][m][e2] + (1.0f - lam) * (float)UB[1][c][m][e2];
      s += uu * (float)Qh[c][m][e2];
    }
    out[c * NB + bb + m] = s;
  }
  if (tid == 0) {
    float s = 0.f;
    for (int m = 0; m < BPB; ++m) s += lamS[0][m];  // reported lam comes from the pos call
    lam_ws[blockIdx.x] = s;
  }
}

__global__ void clockrec_finish(const float* __restrict__ ws,
                                const float* __restrict__ g_tau, const float* __restrict__ e_tau,
                                float* __restrict__ out) {
  __shared__ float red[256];
  int t = threadIdx.x;
  red[t] = ws[t];
  __syncthreads();
  for (int s = 128; s > 0; s >>= 1) { if (t < s) red[t] += red[t + s]; __syncthreads(); }
  if (t == 0) {
    out[2 * NB + 0] = red[0] / (float)NB;
    out[2 * NB + 1] = log1pf(expf(g_tau[0])) + 1e-6f;
    out[2 * NB + 2] = log1pf(expf(e_tau[0])) + 1e-6f;
  }
}

extern "C" void kernel_launch(void* const* d_in, const int* in_sizes, int n_in,
                              void* d_out, int out_size, void* d_ws, size_t ws_size,
                              hipStream_t stream) {
  const int* items_l = (const int*)d_in[0];
  const int* dts_l   = (const int*)d_in[1];
  const unsigned char* mask_l = (const unsigned char*)d_in[2];
  const int* items_s = (const int*)d_in[3];
  const int* dts_s   = (const int*)d_in[4];
  const unsigned char* mask_s = (const unsigned char*)d_in[5];
  const int* pos_it  = (const int*)d_in[6];
  const int* neg_it  = (const int*)d_in[7];
  const float* item_table = (const float*)d_in[8];
  const float* fW1 = (const float*)d_in[9];
  const float* fb1 = (const float*)d_in[10];
  const float* fW2 = (const float*)d_in[11];
  const float* fb2 = (const float*)d_in[12];
  const float* g_dt = (const float*)d_in[13];
  const float* g_cW = (const float*)d_in[14];
  const float* g_cb = (const float*)d_in[15];
  const float* g_kW = (const float*)d_in[16];
  const float* g_kb = (const float*)d_in[17];
  const float* g_vW = (const float*)d_in[18];
  const float* g_vb = (const float*)d_in[19];
  const float* g_tau = (const float*)d_in[20];
  const float* e_dt = (const float*)d_in[21];
  const float* e_cW = (const float*)d_in[22];
  const float* e_cb = (const float*)d_in[23];
  const float* e_kW = (const float*)d_in[24];
  const float* e_kb = (const float*)d_in[25];
  const float* e_vW = (const float*)d_in[26];
  const float* e_vb = (const float*)d_in[27];
  const float* e_tau = (const float*)d_in[28];
  float* out = (float*)d_out;
  float* lam_ws = (float*)d_ws;

  clockrec_main<<<NB / BPB, 512, 0, stream>>>(
      items_l, dts_l, mask_l, items_s, dts_s, mask_s, pos_it, neg_it, item_table,
      fW1, fb1, fW2, fb2,
      g_dt, g_cW, g_cb, g_kW, g_kb, g_vW, g_vb, g_tau,
      e_dt, e_cW, e_cb, e_kW, e_kb, e_vW, e_vb, e_tau,
      out, lam_ws);
  clockrec_finish<<<1, 256, 0, stream>>>(lam_ws, g_tau, e_tau, out);
}
